// MultiHeadAttention_53790170415211
// MI455X (gfx1250) — matrix operand order repaired and checked by reading, not compiled, not run on hardware
//
#include <hip/hip_runtime.h>
#include <hip/hip_bf16.h>

// ---------------------------------------------------------------------------
// CDNA5 (gfx1250) wave32 WMMA types
// ---------------------------------------------------------------------------
typedef __attribute__((ext_vector_type(16))) __bf16 v16bf;
typedef __attribute__((ext_vector_type(8)))  float  v8f;

#define WMMA_BF16(a, b, c) \
  __builtin_amdgcn_wmma_f32_16x16x32_bf16(false, (a), false, (b), (short)0, (c), false, false)

// Async global->LDS path (GLOBAL_LOAD_ASYNC_TO_LDS_B128, ASYNCcnt).
// Probe-discovered signature: (int4 AS1*, int4 AS3*, imm int, imm int)
#if defined(__has_builtin)
#  if __has_builtin(__builtin_amdgcn_global_load_async_to_lds_b128)
#    define HAVE_ASYNC_LDS 1
#  endif
#endif
#ifndef HAVE_ASYNC_LDS
#  define HAVE_ASYNC_LDS 0
#endif

typedef int v4i_vec __attribute__((vector_size(16)));
typedef __attribute__((address_space(1))) v4i_vec* g1_v4i;
typedef __attribute__((address_space(3))) v4i_vec* l3_v4i;

__device__ __forceinline__ void copy16_g2lds(unsigned short* dst_lds,
                                             const unsigned short* src_glb) {
#if HAVE_ASYNC_LDS
  __builtin_amdgcn_global_load_async_to_lds_b128(
      (g1_v4i)src_glb, (l3_v4i)dst_lds, 0, 0);
#else
  *(uint4*)dst_lds = *(const uint4*)src_glb;
#endif
}

__device__ __forceinline__ void wait_async_copies() {
#if HAVE_ASYNC_LDS
#  if defined(__has_builtin) && __has_builtin(__builtin_amdgcn_s_wait_asynccnt)
  __builtin_amdgcn_s_wait_asynccnt(0);
#  else
  asm volatile("s_wait_asynccnt 0" ::: "memory");
#  endif
#endif
}

__device__ __forceinline__ unsigned short f2bf(float f) {
  union { float f; unsigned u; } v; v.f = f;
  unsigned u = v.u;
  unsigned r = u + 0x7FFFu + ((u >> 16) & 1u);   // round-to-nearest-even
  return (unsigned short)(r >> 16);
}

union ABFrag {
  v16bf v;
  unsigned short h[16];
  uint4 q[2];
};

// A fragment (16x32 bf16, M x K), source row-major [m][k], stride ld (elems).
// ISA: lane<16: M=lane, K=0..7 & 16..23 ; lane>=16: M=lane-16, K=8..15 & 24..31
__device__ __forceinline__ v16bf load_frag_a(const unsigned short* p, int ld) {
  int lane = threadIdx.x & 31;
  int row  = lane & 15;
  int k0   = (lane >> 4) << 3;            // 0 or 8
  const unsigned short* r = p + row * ld;
  ABFrag f;
  f.q[0] = *(const uint4*)(r + k0);
  f.q[1] = *(const uint4*)(r + k0 + 16);
  return f.v;
}

// B fragment (32x16 bf16, K x N), source stored TRANSPOSED [n][k], stride ld.
// ISA: lane<16: N=lane, K=0..7 & 16..23 ; lane>=16: N=lane-16, K=8..15 & 24..31
__device__ __forceinline__ v16bf load_frag_b(const unsigned short* p, int ld) {
  int lane = threadIdx.x & 31;
  int n    = lane & 15;
  int k0   = (lane >> 4) << 3;            // 0 or 8
  const unsigned short* r = p + n * ld + k0;
  ABFrag f;
  f.q[0] = *(const uint4*)(r);
  f.q[1] = *(const uint4*)(r + 16);
  return f.v;
}

// ---------------------------------------------------------------------------
// bf16-WMMA GEMM with register-staged software pipelining.
// C[M][N] = A[M][K] * W[K][N] + bias[N]; BM=BN=128, BK=32; 8 waves x (64x32).
// EPI==0: QKV scatter epilogue; EPI==1: plain f32 [m][n].
// ---------------------------------------------------------------------------
template <int EPI, bool A_BF16>
__global__ __launch_bounds__(256) void gemm_wmma(
    const void* __restrict__ Aptr, const float* __restrict__ W,
    const float* __restrict__ bias,
    unsigned short* __restrict__ Qo, unsigned short* __restrict__ Ko,
    unsigned short* __restrict__ Vto, float* __restrict__ Out,
    int M, int N, int Kdim) {
  constexpr int BM = 128, BN = 128, BK = 32, LD = 40;
  __shared__ unsigned short sA[BM * LD];
  __shared__ unsigned short sB[BN * LD];   // transposed: [n][k]

  const int tid = threadIdx.x;
  const int m0 = blockIdx.y * BM;
  const int n0 = blockIdx.x * BN;
  const int lane = tid & 31;
  const int wid = tid >> 5;
  const int waveM = wid >> 2;
  const int waveN = wid & 3;
  const int hl = lane >> 4;
  const int ln = lane & 15;

  // per-thread staging coordinates (4 chunks of 4 elems each for A and B)
  int arow[4], ac4[4], brow[4], bn4[4];
  for (int i = 0; i < 4; ++i) {
    int idx = tid + i * 256;
    arow[i] = idx >> 3;           // 0..127
    ac4[i]  = (idx & 7) << 2;     // 0..28
    brow[i] = idx >> 5;           // 0..31  (k)
    bn4[i]  = (idx & 31) << 2;    // 0..124 (n)
  }

  float4 ra[4];   // A staging (f32 source)
  uint2  rab[4];  // A staging (bf16 source)
  float4 rb[4];   // B staging (f32 weights)

  auto loadA = [&](int kb) {
    if (A_BF16) {
      const unsigned short* Ag = (const unsigned short*)Aptr;
      for (int i = 0; i < 4; ++i)
        rab[i] = *(const uint2*)(Ag + (size_t)(m0 + arow[i]) * Kdim + kb + ac4[i]);
    } else {
      const float* Ag = (const float*)Aptr;
      for (int i = 0; i < 4; ++i)
        ra[i] = *(const float4*)(Ag + (size_t)(m0 + arow[i]) * Kdim + kb + ac4[i]);
    }
  };
  auto loadB = [&](int kb) {
    for (int i = 0; i < 4; ++i)
      rb[i] = *(const float4*)(W + (size_t)(kb + brow[i]) * N + n0 + bn4[i]);
  };
  auto storeA = [&]() {
    if (A_BF16) {
      for (int i = 0; i < 4; ++i)
        *(uint2*)(&sA[arow[i] * LD + ac4[i]]) = rab[i];
    } else {
      for (int i = 0; i < 4; ++i) {
        uint2 p;
        p.x = (unsigned)f2bf(ra[i].x) | ((unsigned)f2bf(ra[i].y) << 16);
        p.y = (unsigned)f2bf(ra[i].z) | ((unsigned)f2bf(ra[i].w) << 16);
        *(uint2*)(&sA[arow[i] * LD + ac4[i]]) = p;
      }
    }
  };
  auto storeB = [&]() {
    for (int i = 0; i < 4; ++i) {
      sB[(bn4[i] + 0) * LD + brow[i]] = f2bf(rb[i].x);
      sB[(bn4[i] + 1) * LD + brow[i]] = f2bf(rb[i].y);
      sB[(bn4[i] + 2) * LD + brow[i]] = f2bf(rb[i].z);
      sB[(bn4[i] + 3) * LD + brow[i]] = f2bf(rb[i].w);
    }
  };

  v8f acc[4][2];
  for (int i = 0; i < 4; ++i)
    for (int j = 0; j < 2; ++j)
      acc[i][j] = v8f{};

  // pipeline prologue: tile 0 into LDS
  loadA(0); loadB(0);
  storeA(); storeB();
  __syncthreads();

  for (int kb = 0; kb < Kdim; kb += BK) {
    const bool hasNext = (kb + BK) < Kdim;
    if (hasNext) { loadA(kb + BK); loadB(kb + BK); }   // overlap with compute

    v16bf af[4], bfr[2];
    for (int i = 0; i < 4; ++i)
      af[i] = load_frag_a(&sA[(waveM * 64 + i * 16) * LD], LD);
    for (int j = 0; j < 2; ++j)
      bfr[j] = load_frag_b(&sB[(waveN * 32 + j * 16) * LD], LD);
    for (int i = 0; i < 4; ++i)
      for (int j = 0; j < 2; ++j)
        acc[i][j] = WMMA_BF16(af[i], bfr[j], acc[i][j]);

    __syncthreads();
    if (hasNext) { storeA(); storeB(); }
    __syncthreads();
  }

  // ---- epilogue ----
  for (int i = 0; i < 4; ++i) {
    for (int j = 0; j < 2; ++j) {
      for (int v = 0; v < 8; ++v) {
        int m = m0 + waveM * 64 + i * 16 + v + hl * 8;
        int n = n0 + waveN * 32 + j * 16 + ln;
        float val = acc[i][j][v] + bias[n];
        if (EPI == 0) {
          int b = m >> 11, s = m & 2047;
          int h = n / 192, r = n - h * 192;
          unsigned short bv = f2bf(val);
          size_t headBase = ((size_t)(b * 16 + h)) * 2048;
          if (r < 64) {
            Qo[(headBase + s) * 64 + r] = bv;
          } else if (r < 128) {
            Ko[(headBase + s) * 64 + (r - 64)] = bv;
          } else {
            Vto[((size_t)(b * 16 + h) * 64 + (r - 128)) * 2048 + s] = bv;
          }
        } else {
          Out[(size_t)m * 1024 + n] = val;
        }
      }
    }
  }
}

// ---------------------------------------------------------------------------
// Flash attention: grid = (S/64 query tiles, B*H). 128 threads = 4 waves;
// each wave owns 16 query rows. Key blocks of 64, dh = 64.
// ---------------------------------------------------------------------------
__global__ __launch_bounds__(128) void flash_attn(
    const unsigned short* __restrict__ Q, const unsigned short* __restrict__ K,
    const unsigned short* __restrict__ Vt, const float* __restrict__ mask,
    unsigned short* __restrict__ concat) {
  constexpr int S = 2048, H = 16, LD = 72;
  const int bh = blockIdx.y;
  const int b = bh >> 4, h = bh & 15;
  const int q0 = blockIdx.x * 64;
  const int tid = threadIdx.x;
  const int lane = tid & 31, wid = tid >> 5;
  const int hl = lane >> 4, ln = lane & 15;

  __shared__ unsigned short sK[64 * LD];        // [key][dh]
  __shared__ unsigned short sV[64 * LD];        // [dh][key]  (from Vt)
  __shared__ unsigned short sP[4][16 * LD];     // per-wave P: [qrow][key]

  const unsigned short* Qbh = Q  + (size_t)(b * H + h) * S * 64;
  const unsigned short* Kbh = K  + (size_t)(b * H + h) * S * 64;
  const unsigned short* Vbh = Vt + (size_t)(b * H + h) * 64 * S;
  const float* mbase = mask + (size_t)b * S;

  // Q fragments for this wave's 16 rows (2 K-steps of 32 over dh=64)
  v16bf qf[2];
  {
    const unsigned short* qp = Qbh + (size_t)(q0 + wid * 16 + ln) * 64;
    for (int s = 0; s < 2; ++s) {
      int k0 = s * 32 + hl * 8;
      ABFrag f;
      f.q[0] = *(const uint4*)(qp + k0);
      f.q[1] = *(const uint4*)(qp + k0 + 16);
      qf[s] = f.v;
    }
  }

  float mrow[8], lrow[8];
  v8f accO[4];
  for (int v = 0; v < 8; ++v) { mrow[v] = -1e30f; lrow[v] = 0.0f; }
  for (int nf = 0; nf < 4; ++nf) accO[nf] = v8f{};

  for (int kb = 0; kb < S; kb += 64) {
    // cooperative async load of K tile [key][dh] and V tile [dh][key]
    for (int i = 0; i < 4; ++i) {
      int idx = tid + i * 128;              // 512 chunks of 8 bf16
      int row = idx >> 3;                   // 0..63
      int c8  = (idx & 7) << 3;             // 0..56
      copy16_g2lds(&sK[row * LD + c8], Kbh + (size_t)(kb + row) * 64 + c8);
      copy16_g2lds(&sV[row * LD + c8], Vbh + (size_t)row * S + kb + c8);
    }
    if (kb + 64 < S) {
      __builtin_prefetch(Kbh + (size_t)(kb + 64 + (tid >> 1)) * 64, 0, 0);
      __builtin_prefetch(Vbh + (size_t)(tid >> 1) * S + kb + 64, 0, 0);
    }
    wait_async_copies();
    __syncthreads();

    // ---- logits: preload all 8 B fragments, then 8 WMMAs
    v16bf bk[4][2];
    for (int t = 0; t < 4; ++t) {
      bk[t][0] = load_frag_b(&sK[(t * 16) * LD], LD);       // dh 0..31
      bk[t][1] = load_frag_b(&sK[(t * 16) * LD + 32], LD);  // dh 32..63
    }
    v8f cl[4];
    for (int t = 0; t < 4; ++t) {
      v8f c = v8f{};
      c = WMMA_BF16(qf[0], bk[t][0], c);
      c = WMMA_BF16(qf[1], bk[t][1], c);
      cl[t] = c;
    }
    for (int t = 0; t < 4; ++t) {
      int key = kb + t * 16 + ln;
      float mk = mbase[key] * -1e9f;
      for (int v = 0; v < 8; ++v) cl[t][v] = cl[t][v] * 0.125f + mk;
    }

    // ---- online softmax (row = VGPR index within a 16-lane half)
    float mnew[8], sc[8];
    for (int v = 0; v < 8; ++v) {
      float mx = fmaxf(fmaxf(cl[0][v], cl[1][v]), fmaxf(cl[2][v], cl[3][v]));
      for (int off = 1; off < 16; off <<= 1)
        mx = fmaxf(mx, __shfl_xor(mx, off, 32));
      mnew[v] = fmaxf(mrow[v], mx);
      sc[v] = __expf(mrow[v] - mnew[v]);
      mrow[v] = mnew[v];
    }
    for (int t = 0; t < 4; ++t)
      for (int v = 0; v < 8; ++v)
        cl[t][v] = __expf(cl[t][v] - mnew[v]);
    for (int v = 0; v < 8; ++v) {
      float s = cl[0][v] + cl[1][v] + cl[2][v] + cl[3][v];
      for (int off = 1; off < 16; off <<= 1)
        s += __shfl_xor(s, off, 32);
      lrow[v] = lrow[v] * sc[v] + s;
    }
    for (int nf = 0; nf < 4; ++nf)
      for (int v = 0; v < 8; ++v)
        accO[nf][v] *= sc[v];

    // ---- transpose P via per-wave LDS: (row = v + hl*8, key = t*16+ln)
    unsigned short* Pw = &sP[wid][0];
    for (int t = 0; t < 4; ++t)
      for (int v = 0; v < 8; ++v)
        Pw[(v + hl * 8) * LD + t * 16 + ln] = f2bf(cl[t][v]);
    __syncthreads();

    // ---- PV: preload fragments, then 8 WMMAs
    v16bf ap[2];
    ap[0] = load_frag_a(&Pw[0], LD);
    ap[1] = load_frag_a(&Pw[32], LD);
    v16bf bv[4][2];
    for (int nf = 0; nf < 4; ++nf) {
      bv[nf][0] = load_frag_b(&sV[(nf * 16) * LD], LD);
      bv[nf][1] = load_frag_b(&sV[(nf * 16) * LD + 32], LD);
    }
    for (int nf = 0; nf < 4; ++nf) {
      accO[nf] = WMMA_BF16(ap[0], bv[nf][0], accO[nf]);
      accO[nf] = WMMA_BF16(ap[1], bv[nf][1], accO[nf]);
    }
    __syncthreads();
  }

  // ---- finalize: O /= l, write bf16 concat (B, S, H*dh)
  for (int nf = 0; nf < 4; ++nf) {
    for (int v = 0; v < 8; ++v) {
      int m = q0 + wid * 16 + v + hl * 8;
      float o = accO[nf][v] / lrow[v];
      int dcol = h * 64 + nf * 16 + ln;
      concat[((size_t)b * S + m) * 1024 + dcol] = f2bf(o);
    }
  }
}

// ---------------------------------------------------------------------------
// Host launcher
// ---------------------------------------------------------------------------
extern "C" void kernel_launch(void* const* d_in, const int* in_sizes, int n_in,
                              void* d_out, int out_size, void* d_ws, size_t ws_size,
                              hipStream_t stream) {
  const float* x    = (const float*)d_in[0];   // (4, 2048, 1024)
  const float* Wqkv = (const float*)d_in[1];   // (1024, 3072)
  const float* bqkv = (const float*)d_in[2];   // (3072,)
  const float* Wo   = (const float*)d_in[3];   // (1024, 1024)
  const float* bo   = (const float*)d_in[4];   // (1024,)
  const float* mask = (const float*)d_in[5];   // (4, 1, 1, 2048)
  float* out = (float*)d_out;                  // (4, 2048, 1024) f32

  const size_t HEADS_ELEMS = (size_t)4 * 16 * 2048 * 64;   // 8M bf16 elems
  unsigned short* Q  = (unsigned short*)d_ws;
  unsigned short* Kb = Q  + HEADS_ELEMS;
  unsigned short* Vt = Kb + HEADS_ELEMS;
  unsigned short* Cc = Vt + HEADS_ELEMS;       // concat (B,S,D) bf16

  gemm_wmma<0, false><<<dim3(3072 / 128, 8192 / 128), 256, 0, stream>>>(
      x, Wqkv, bqkv, Q, Kb, Vt, nullptr, 8192, 3072, 1024);

  flash_attn<<<dim3(2048 / 64, 4 * 16), 128, 0, stream>>>(Q, Kb, Vt, mask, Cc);

  gemm_wmma<1, true><<<dim3(1024 / 128, 8192 / 128), 256, 0, stream>>>(
      Cc, Wo, bo, nullptr, nullptr, nullptr, out, 8192, 1024, 1024);
}
